// complex_HGRN_58153857187912
// MI455X (gfx1250) — matrix operand order repaired and checked by reading, not compiled, not run on hardware
//
#include <hip/hip_runtime.h>
#include <hip/hip_bf16.h>

// D-MPNN (Chemprop-style) forward for MI455X / gfx1250.
// GEMMs on v_wmma_f32_16x16x32_bf16 (fp32 accumulate); activations/aggregates
// fp32. segment_sum fused into each GEMM epilogue via global_atomic_add_f32.
// All hot-path indexing is 32-bit (buffers < 4GB) so the backend can use
// SGPR-base + 32-bit voffset global addressing.

#define N_NODES 50000
#define N_EDGES 800000
#define HIDDEN  128

typedef __attribute__((ext_vector_type(16))) __bf16 v16bf;
typedef __attribute__((ext_vector_type(8)))  float  v8f;

static __device__ __forceinline__ v8f wmma_bf16(v16bf a, v16bf b, v8f c) {
  // 8 args: (neg_a, A, neg_b, B, c_mod, C, reuse_a, reuse_b)
  return __builtin_amdgcn_wmma_f32_16x16x32_bf16(false, a, false, b,
                                                 (short)0, c, false, false);
}

// load 8 contiguous fp32 (32B) via two b128 loads; 32-bit element offset
static __device__ __forceinline__ void ld8(const float* __restrict__ p,
                                           unsigned off, float* d) {
  float4 a = *(const float4*)(p + off);
  float4 b = *(const float4*)(p + off + 4);
  d[0]=a.x; d[1]=a.y; d[2]=a.z; d[3]=a.w;
  d[4]=b.x; d[5]=b.y; d[6]=b.z; d[7]=b.w;
}

// A-fragment packing (ISA 16-bit A 16x32 layout):
// lane holds row M=lane&15; K = 8*hi+i (elems 0..7) and 16+8*hi+i (elems 8..15)
static __device__ __forceinline__ v16bf pack_a(const float* c1, const float* c2) {
  v16bf a;
#pragma unroll
  for (int i = 0; i < 8; ++i) { a[i] = (__bf16)c1[i]; a[8+i] = (__bf16)c2[i]; }
  return a;
}

// B-fragment: two 8-bf16 chunks at off and off+16 from LDS (two ds_read_b128)
static __device__ __forceinline__ v16bf ldb(const __bf16* __restrict__ w,
                                            unsigned off) {
  union { uint4 q[2]; v16bf v; } u;
  u.q[0] = *(const uint4*)(w + off);
  u.q[1] = *(const uint4*)(w + off + 16);
  return u.v;
}

// B-fragment straight from global fp32 (cache-resident weights)
static __device__ __forceinline__ v16bf ldb_g(const float* __restrict__ w,
                                              unsigned off) {
  float c[16];
  ld8(w, off, c); ld8(w, off + 16, c + 8);
  v16bf b;
#pragma unroll
  for (int i = 0; i < 16; ++i) b[i] = (__bf16)c[i];
  return b;
}

// convert OUTxK fp32 weights to bf16 in LDS, rows padded to KP (KP % 8 == 0)
static __device__ __forceinline__ void fill_w_lds(__bf16* wl,
                                                  const float* __restrict__ W,
                                                  int K, int KP) {
  for (int i = threadIdx.x; i < HIDDEN * KP; i += blockDim.x) {
    int o = i / KP, k = i - o * KP;
    wl[i] = (__bf16)((k < K) ? W[o * K + k] : 0.f);
  }
  __syncthreads();
}

// ---------------------------------------------------------------------------
// H0 = ReLU(W_i [x[src]; edge_attr] + b_i); fused scatter-add into agg0
// ---------------------------------------------------------------------------
__global__ __launch_bounds__(256) void edge_init_kernel(
    const float* __restrict__ x, const float* __restrict__ ea,
    const float* __restrict__ Wi, const float* __restrict__ bi,
    const int* __restrict__ src, const int* __restrict__ dst,
    float* __restrict__ H0, float* __restrict__ agg0)
{
  const unsigned KP = 168;                  // K=144 padded (16B-aligned rows)
  extern __shared__ __align__(16) char smem[];
  __bf16* wl = (__bf16*)smem;
  fill_w_lds(wl, Wi, 144, KP);

  const unsigned lane = threadIdx.x & 31;
  const unsigned n16  = lane & 15;
  const unsigned hi   = lane >> 4;
  const unsigned tile = blockIdx.x * 8 + (threadIdx.x >> 5);
  const unsigned e_base = tile * 16;
  if (e_base >= N_EDGES) return;

  const unsigned e_lane = e_base + n16;
  const unsigned s = (unsigned)src[e_lane];

  v16bf A[5];
  {
    float c1[8], c2[8];
#pragma unroll
    for (int t = 0; t < 4; ++t) {
      ld8(x, s * 128u + t*32 + hi*8,      c1);
      ld8(x, s * 128u + t*32 + 16 + hi*8, c2);
      A[t] = pack_a(c1, c2);
    }
    ld8(ea, e_lane * 16u + hi*8, c1);           // bond features, K 128..143
#pragma unroll
    for (int i = 0; i < 8; ++i) c2[i] = 0.f;    // zero-pad K 144..159
    A[4] = pack_a(c1, c2);
  }

  unsigned dste[8];
#pragma unroll
  for (int r = 0; r < 8; ++r) dste[r] = (unsigned)dst[e_base + r + 8*hi];

  const unsigned brow = n16 * KP + hi * 8;

#pragma unroll
  for (unsigned o = 0; o < 8; ++o) {
    v16bf B[5];
#pragma unroll
    for (int t = 0; t < 5; ++t)                 // all ds_loads in flight first
      B[t] = ldb(wl, (o*16) * KP + brow + t*32);
    v8f acc = {};
#pragma unroll
    for (int t = 0; t < 5; ++t)                 // back-to-back WMMA chain
      acc = wmma_bf16(A[t], B[t], acc);

    const float bias = bi[o*16 + n16];
#pragma unroll
    for (unsigned r = 0; r < 8; ++r) {
      const unsigned e = e_base + r + 8*hi;
      const unsigned idx = e * (unsigned)HIDDEN + o*16 + n16;
      float v = acc[r] + bias;
      v = v > 0.f ? v : 0.f;
      H0[idx] = v;
      atomicAdd(agg0 + (dste[r] * (unsigned)HIDDEN + o*16 + n16), v);
    }
  }
}

// ---------------------------------------------------------------------------
// Hout = ReLU(H0 + W_h (aggIn[src] - Hprev[rev]) + b_h); scatter-add -> aggOut
// Safe in-place (Hout == Hprev): rev = e^1 stays inside each 16-edge tile and
// all A-fragments are cached in registers before any store.
// ---------------------------------------------------------------------------
__global__ __launch_bounds__(256) void edge_layer_kernel(
    const float* __restrict__ Hprev, const float* __restrict__ H0,
    float* __restrict__ Hout,
    const float* __restrict__ aggIn, float* __restrict__ aggOut,
    const float* __restrict__ Wh, const float* __restrict__ bh,
    const int* __restrict__ src, const int* __restrict__ dst,
    const int* __restrict__ rev)
{
  const unsigned KP = 136;                  // K=128 padded
  extern __shared__ __align__(16) char smem[];
  __bf16* wl = (__bf16*)smem;
  fill_w_lds(wl, Wh, 128, KP);

  const unsigned lane = threadIdx.x & 31;
  const unsigned n16  = lane & 15;
  const unsigned hi   = lane >> 4;
  const unsigned tile = blockIdx.x * 8 + (threadIdx.x >> 5);
  const unsigned e_base = tile * 16;
  if (e_base >= N_EDGES) return;

  const unsigned e_lane = e_base + n16;
  const unsigned s  = (unsigned)src[e_lane];
  const unsigned re = (unsigned)rev[e_lane];

  v16bf A[4];
  {
    float c1[8], c2[8], d1[8], d2[8];
#pragma unroll
    for (int t = 0; t < 4; ++t) {
      ld8(aggIn, s  * 128u + t*32 + hi*8,      c1);
      ld8(aggIn, s  * 128u + t*32 + 16 + hi*8, c2);
      ld8(Hprev, re * 128u + t*32 + hi*8,      d1);
      ld8(Hprev, re * 128u + t*32 + 16 + hi*8, d2);
#pragma unroll
      for (int i = 0; i < 8; ++i) { c1[i] -= d1[i]; c2[i] -= d2[i]; }
      A[t] = pack_a(c1, c2);
    }
  }

  unsigned dste[8];
#pragma unroll
  for (int r = 0; r < 8; ++r) dste[r] = (unsigned)dst[e_base + r + 8*hi];

  const unsigned brow = n16 * KP + hi * 8;

#pragma unroll
  for (unsigned o = 0; o < 8; ++o) {
    v16bf B[4];
#pragma unroll
    for (int t = 0; t < 4; ++t)
      B[t] = ldb(wl, (o*16) * KP + brow + t*32);
    v8f acc = {};
#pragma unroll
    for (int t = 0; t < 4; ++t)
      acc = wmma_bf16(A[t], B[t], acc);

    const float bias = bh[o*16 + n16];
#pragma unroll
    for (unsigned r = 0; r < 8; ++r) {
      const unsigned e = e_base + r + 8*hi;
      const unsigned idx = e * (unsigned)HIDDEN + o*16 + n16;
      float v = acc[r] + bias + H0[idx];
      v = v > 0.f ? v : 0.f;
      Hout[idx] = v;
      atomicAdd(aggOut + (dste[r] * (unsigned)HIDDEN + o*16 + n16), v);
    }
  }
}

// ---------------------------------------------------------------------------
// isolated-node fallback: agg[n] = (rowsum(agg[n]) == 0) ? x[n] : agg[n]
// ---------------------------------------------------------------------------
__global__ __launch_bounds__(128) void select_kernel(
    const float* __restrict__ x, float* __restrict__ agg)
{
  const unsigned node = blockIdx.x;
  const unsigned t = threadIdx.x;
  const unsigned idx = node * (unsigned)HIDDEN + t;
  float v = agg[idx];
  float s = v;
#pragma unroll
  for (int off = 16; off > 0; off >>= 1) s += __shfl_down(s, off, 32);
  __shared__ float red[4];
  if ((t & 31) == 0) red[t >> 5] = s;
  __syncthreads();
  const float tot = red[0] + red[1] + red[2] + red[3];
  if (tot == 0.f) agg[idx] = x[idx];
}

// ---------------------------------------------------------------------------
// out = ReLU(W_o [x ; M] + b_o)   (K = 256; weights served from cache)
// ---------------------------------------------------------------------------
__global__ __launch_bounds__(256) void readout_kernel(
    const float* __restrict__ x, const float* __restrict__ M,
    const float* __restrict__ Wo, const float* __restrict__ bo,
    float* __restrict__ out)
{
  const unsigned lane = threadIdx.x & 31;
  const unsigned n16  = lane & 15;
  const unsigned hi   = lane >> 4;
  const unsigned tile = blockIdx.x * 8 + (threadIdx.x >> 5);
  const unsigned n_tiles = N_NODES / 16;
  if (tile >= n_tiles) return;
  const unsigned nb = tile * 16;
  const unsigned node = nb + n16;

  v16bf A[8];
  {
    float c1[8], c2[8];
#pragma unroll
    for (int t = 0; t < 4; ++t) {
      ld8(x, node * 128u + t*32 + hi*8,      c1);
      ld8(x, node * 128u + t*32 + 16 + hi*8, c2);
      A[t] = pack_a(c1, c2);
    }
#pragma unroll
    for (int t = 0; t < 4; ++t) {
      ld8(M, node * 128u + t*32 + hi*8,      c1);
      ld8(M, node * 128u + t*32 + 16 + hi*8, c2);
      A[4 + t] = pack_a(c1, c2);
    }
  }

#pragma unroll
  for (unsigned o = 0; o < 8; ++o) {
    v8f acc = {};
#pragma unroll
    for (unsigned t = 0; t < 8; ++t) {
      v16bf B = ldb_g(Wo, (o*16 + n16) * 256u + t*32 + hi*8);
      acc = wmma_bf16(A[t], B, acc);
    }
    const float bias = bo[o*16 + n16];
#pragma unroll
    for (unsigned r = 0; r < 8; ++r) {
      const unsigned idx = (nb + r + 8*hi) * (unsigned)HIDDEN + o*16 + n16;
      float v = acc[r] + bias;
      out[idx] = v > 0.f ? v : 0.f;
    }
  }
}

// ---------------------------------------------------------------------------
extern "C" void kernel_launch(void* const* d_in, const int* in_sizes, int n_in,
                              void* d_out, int out_size, void* d_ws, size_t ws_size,
                              hipStream_t stream) {
  (void)in_sizes; (void)n_in; (void)out_size; (void)ws_size;

  const float* x   = (const float*)d_in[0];   // [N,128]
  const float* ea  = (const float*)d_in[1];   // [E,16]
  const float* Wi  = (const float*)d_in[2];   // [128,144]
  const float* bi  = (const float*)d_in[3];   // [128]
  const float* Wh  = (const float*)d_in[4];   // [128,128]
  const float* bh  = (const float*)d_in[5];   // [128]
  const float* Wo  = (const float*)d_in[6];   // [128,256]
  const float* bo  = (const float*)d_in[7];   // [128]
  const int*   ei  = (const int*)d_in[8];     // [2,E]
  const int*   rev = (const int*)d_in[9];     // [E]
  float* out = (float*)d_out;

  const int* src = ei;             // edge_index[0]
  const int* dst = ei + N_EDGES;   // edge_index[1]

  // workspace: agg0 | agg1 | agg2 | H0 | H   (~896 MB fp32)
  float* agg0 = (float*)d_ws;
  float* agg1 = agg0 + (size_t)N_NODES * HIDDEN;
  float* agg2 = agg1 + (size_t)N_NODES * HIDDEN;
  float* H0   = agg2 + (size_t)N_NODES * HIDDEN;
  float* H    = H0   + (size_t)N_EDGES * HIDDEN;

  hipMemsetAsync(agg0, 0, (size_t)3 * N_NODES * HIDDEN * sizeof(float), stream);

  const dim3 blk(256);
  const int edge_blocks = (N_EDGES / 16) / 8;               // 6250, exact
  const int node_blocks = (N_NODES / 16 + 7) / 8;           // 391

  const size_t lds_init  = (size_t)HIDDEN * 168 * sizeof(__bf16); // 43008 B
  const size_t lds_layer = (size_t)HIDDEN * 136 * sizeof(__bf16); // 34816 B

  // H0 + segment_sum(H0) -> agg0
  edge_init_kernel<<<edge_blocks, blk, lds_init, stream>>>(
      x, ea, Wi, bi, src, dst, H0, agg0);

  // layer 1: H1 from H0/agg0, scatter -> agg1
  edge_layer_kernel<<<edge_blocks, blk, lds_layer, stream>>>(
      H0, H0, H, agg0, agg1, Wh, bh, src, dst, rev);

  // layer 2 (in-place on H): H2 from H1/agg1, scatter -> agg2
  edge_layer_kernel<<<edge_blocks, blk, lds_layer, stream>>>(
      H, H0, H, agg1, agg2, Wh, bh, src, dst, rev);

  // isolated-node fallback into agg2
  select_kernel<<<N_NODES, 128, 0, stream>>>(x, agg2);

  // readout
  readout_kernel<<<node_blocks, blk, 0, stream>>>(x, agg2, Wo, bo, out);
}
